// GAT_encoder_14130442403921
// MI455X (gfx1250) — matrix operand order repaired and checked by reading, not compiled, not run on hardware
//
#include <hip/hip_runtime.h>
#include <hip/hip_bf16.h>
#include <stdint.h>

typedef __attribute__((ext_vector_type(16))) _Float16 v16h;
typedef __attribute__((ext_vector_type(8)))  float    v8f;

#define HID 128
#define HEADS 4
#define OUTD 256
#define NEG_SLOPE 0.2f

// ---------------------------------------------------------------------------
// CDNA5 async copy: global -> LDS, 16 B per lane, tracked by ASYNCcnt.
// (inline asm: portable across ROCm 7.2 / amdgpu-toolchain builtin arities)
// ---------------------------------------------------------------------------
__device__ static inline void async_b128(unsigned lds, const void* g) {
  asm volatile("global_load_async_to_lds_b128 %0, %1, off"
               :: "v"(lds), "v"(g) : "memory");
}
__device__ static inline void async_b128_o16(unsigned lds, const void* g) {
  asm volatile("global_load_async_to_lds_b128 %0, %1, off offset:16"
               :: "v"(lds), "v"(g) : "memory");
}
__device__ static inline void wait_async0() {
  asm volatile("s_wait_asynccnt 0x0" ::: "memory");
}
__device__ static inline unsigned lds_off(const void* p) {
  return (unsigned)(uintptr_t)p;   // low 32 bits of generic addr = LDS offset
}

// ---------------------------------------------------------------------------
// utility kernels (vectorized: all sizes here are multiples of 4)
// ---------------------------------------------------------------------------
__global__ void k_fill_zero4(float4* __restrict__ p, size_t n4) {
  size_t i = (size_t)blockIdx.x * blockDim.x + threadIdx.x;
  if (i < n4) {
    float4 z; z.x = 0.f; z.y = 0.f; z.z = 0.f; z.w = 0.f;
    p[i] = z;
  }
}

__global__ void k_cvt_f16x4(const float4* __restrict__ in, uint2* __restrict__ out, size_t n4) {
  size_t i = (size_t)blockIdx.x * blockDim.x + threadIdx.x;
  if (i < n4) {
    const float4 v = in[i];
    union { _Float16 h[4]; uint2 u; } r;
    r.h[0] = (_Float16)v.x; r.h[1] = (_Float16)v.y;
    r.h[2] = (_Float16)v.z; r.h[3] = (_Float16)v.w;
    out[i] = r.u;
  }
}

// W[K][N] f32 -> Wt[N][K] f16 (one-time, tiny; writes coalesced along k)
__global__ void k_cvt_transpose_f16(const float* __restrict__ in,
                                    _Float16* __restrict__ out, int K, int N) {
  const int i = blockIdx.x * blockDim.x + threadIdx.x;
  if (i >= K * N) return;
  const int n = i / K;
  const int k = i - n * K;
  out[(size_t)n * K + k] = (_Float16)in[(size_t)k * N + n];
}

// ---------------------------------------------------------------------------
// WMMA GEMM: C[M,Ncols] = A[M,K](f16, row-major) x Bt[Ncols,K](f16, N-major).
// Block = 256 threads = 8 waves; block tile 128x64; each wave owns a 16-row
// strip and all four 16-col subtiles -> 4 x v_wmma_f32_16x16x32_f16 per
// 32-wide k-step. Both tiles staged by global_load_async_to_lds_b128 into
// double-buffered LDS (copy of tile t+1 overlaps compute of tile t; ASYNCcnt
// fenced). Zero DS stores: Bt is pre-transposed, so both fragments read as
// contiguous ds_load_b128 pairs. K % 32 == 0, Ncols % 64 == 0; M handled by
// row clamping (garbage rows never stored).
// ---------------------------------------------------------------------------
#define TM 128
#define TN 64
#define TK 32
#define LDS_S 40  // padded row stride in halves (80 B, keeps 16 B align)

__global__ __launch_bounds__(256) void k_gemm_wmma(
    const _Float16* __restrict__ A, const _Float16* __restrict__ Bt,
    float* __restrict__ C, int M, int K, int Ncols)
{
  __shared__ __align__(16) _Float16 As[2][TM][LDS_S];  // [m][k]
  __shared__ __align__(16) _Float16 Bs[2][TN][LDS_S];  // [n][k]

  const int tid  = threadIdx.x;
  const int lane = tid & 31;
  const int wm   = tid >> 5;   // wave 0..7 -> 16-row strip
  const int blockM = blockIdx.y * TM;
  const int blockN = blockIdx.x * TN;

  v8f acc0 = {}, acc1 = {}, acc2 = {}, acc3 = {};

  const int hi = lane >> 4;    // lane group: 0 = lanes 0-15, 1 = lanes 16-31
  const int rl = lane & 15;

  // A staging: 128 rows x 32 halves; 2 threads/row, 2x b128 each (clamped row)
  const int arow  = tid >> 1;
  const int acol  = (tid & 1) * 16;
  const int agrow = blockM + arow;
  const size_t aRowOff = (size_t)(agrow < M ? agrow : M - 1) * K;
  // B staging: 64 n-rows x 32 halves; 4 threads/row, 1x b128 each
  const int brow = tid >> 2;
  const int bcol = (tid & 3) * 8;
  const size_t bRowOff = (size_t)(blockN + brow) * K;

  const unsigned ldsA[2] = { lds_off(&As[0][arow][acol]), lds_off(&As[1][arow][acol]) };
  const unsigned ldsB[2] = { lds_off(&Bs[0][brow][bcol]), lds_off(&Bs[1][brow][bcol]) };

  const int nT = K / TK;
  // prologue: stage tile 0 into buffer 0
  {
    const _Float16* ga = A + aRowOff;
    async_b128(ldsA[0], ga);
    async_b128_o16(ldsA[0], ga);
    async_b128(ldsB[0], Bt + bRowOff);
  }

  for (int t = 0; t < nT; ++t) {
    wait_async0();
    __syncthreads();                       // tile t resident for all waves
    const int cur = t & 1;
    if (t + 1 < nT) {                      // overlap: stage tile t+1
      const int k1 = (t + 1) * TK;
      const _Float16* ga = A + aRowOff + k1;
      async_b128(ldsA[cur ^ 1], ga);
      async_b128_o16(ldsA[cur ^ 1], ga);
      async_b128(ldsB[cur ^ 1], Bt + bRowOff + k1);
    }

    // A fragment (ISA 16-bit 16x32 layout): lanes 0-15 hold K 0-7 / 16-23,
    // lanes 16-31 hold K 8-15 / 24-31, row M = rl of this wave's strip.
    v16h af;
    {
      const int ar = wm * 16 + rl;
      const int kb = hi * 8;
      #pragma unroll
      for (int j = 0; j < 8; ++j) {
        af[j]     = As[cur][ar][kb + j];
        af[8 + j] = As[cur][ar][16 + kb + j];
      }
    }
    // B fragments: col N = rl of each 16-wide subtile; lane group selects
    // K 0-7 / 16-23 vs 8-15 / 24-31 (matches A; ds_load_b128 pairs). Load all four
    // before the WMMAs so dscnt waits can be partial.
    {
      const int kb = hi * 8;
      v16h bf0, bf1, bf2, bf3;
      #pragma unroll
      for (int j = 0; j < 16; ++j) {
        bf0[j] = Bs[cur][rl][(j < 8 ? kb + j : 8 + kb + j)];
        bf1[j] = Bs[cur][16 + rl][(j < 8 ? kb + j : 8 + kb + j)];
        bf2[j] = Bs[cur][32 + rl][(j < 8 ? kb + j : 8 + kb + j)];
        bf3[j] = Bs[cur][48 + rl][(j < 8 ? kb + j : 8 + kb + j)];
      }
      acc0 = __builtin_amdgcn_wmma_f32_16x16x32_f16(false, af, false, bf0, (short)0, acc0, false, false);
      acc1 = __builtin_amdgcn_wmma_f32_16x16x32_f16(false, af, false, bf1, (short)0, acc1, false, false);
      acc2 = __builtin_amdgcn_wmma_f32_16x16x32_f16(false, af, false, bf2, (short)0, acc2, false, false);
      acc3 = __builtin_amdgcn_wmma_f32_16x16x32_f16(false, af, false, bf3, (short)0, acc3, false, false);
    }
    __syncthreads();                       // reads done before buffer reuse
  }

  // C/D layout: VGPR r -> row = r + 8*laneGroup, col = lane%16
  const int rbase = blockM + wm * 16 + hi * 8;
  const int c0    = blockN + rl;
  #pragma unroll
  for (int r = 0; r < 8; ++r) {
    const int row = rbase + r;
    if (row < M) {
      float* cr = C + (size_t)row * Ncols + c0;
      cr[0]  = acc0[r];
      cr[16] = acc1[r];
      cr[32] = acc2[r];
      cr[48] = acc3[r];
    }
  }
}

// ---------------------------------------------------------------------------
// per-(node,head) attention scores: el = <feat, al>, er = <feat, ar>
// one wave32 per (node,head)
// ---------------------------------------------------------------------------
__global__ __launch_bounds__(256) void k_attn_scores(
    const float* __restrict__ feat, const float* __restrict__ al,
    const float* __restrict__ ar, float* __restrict__ el, float* __restrict__ er,
    int Nn, int D)
{
  const int w    = (int)((blockIdx.x * blockDim.x + threadIdx.x) >> 5);
  const int lane = threadIdx.x & 31;
  if (w >= Nn * HEADS) return;
  const int n = w / HEADS;
  const int h = w - n * HEADS;
  const float* f  = feat + ((size_t)n * HEADS + h) * D;
  const float* a0 = al + (size_t)h * D;
  const float* a1 = ar + (size_t)h * D;
  float sl = 0.f, sr = 0.f;
  for (int d = lane; d < D; d += 32) {
    const float v = f[d];
    sl += v * a0[d];
    sr += v * a1[d];
  }
  #pragma unroll
  for (int off = 16; off > 0; off >>= 1) {
    sl += __shfl_xor(sl, off, 32);
    sr += __shfl_xor(sr, off, 32);
  }
  if (lane == 0) { el[w] = sl; er[w] = sr; }
}

// monotone float <-> ordered-uint mapping (so atomic max works on uint)
__device__ inline unsigned f2ord(float f) {
  unsigned u = __float_as_uint(f);
  return u ^ ((unsigned)((int)u >> 31) | 0x80000000u);
}
__device__ inline float ord2f(unsigned u) {
  unsigned b = (u & 0x80000000u) ? (u ^ 0x80000000u) : ~u;
  return __uint_as_float(b);
}

// edge pass A: e = leakyrelu(el[src]+er[dst]); segment-max via atomic max
__global__ void k_edge_max(const int* __restrict__ src, const int* __restrict__ dst,
                           const float* __restrict__ el, const float* __restrict__ er,
                           float* __restrict__ e_edge, unsigned* __restrict__ mOrd,
                           int Etot)
{
  const int i = blockIdx.x * blockDim.x + threadIdx.x;
  if (i >= Etot * HEADS) return;
  const int e = i >> 2;
  const int h = i & 3;
  float v = el[src[e] * HEADS + h] + er[dst[e] * HEADS + h];
  v = (v > 0.f) ? v : NEG_SLOPE * v;
  e_edge[i] = v;
  atomicMax(&mOrd[dst[e] * HEADS + h], f2ord(v));
}

// edge pass B: ex = exp(e - m[dst]); segment-sum via atomic add; e_edge := ex
__global__ void k_edge_exp(const int* __restrict__ dst, float* __restrict__ e_edge,
                           const unsigned* __restrict__ mOrd, float* __restrict__ s,
                           int Etot)
{
  const int i = blockIdx.x * blockDim.x + threadIdx.x;
  if (i >= Etot * HEADS) return;
  const int e  = i >> 2;
  const int h  = i & 3;
  const int di = dst[e] * HEADS + h;
  const float ex = __expf(e_edge[i] - ord2f(mOrd[di]));
  e_edge[i] = ex;
  atomicAdd(&s[di], ex);
}

// edge pass C: rst[dst] += alpha * feat[src] (block per edge, L2 f32 atomics)
__global__ __launch_bounds__(256) void k_edge_aggregate(
    const int* __restrict__ src, const int* __restrict__ dst,
    const float* __restrict__ e_edge, const float* __restrict__ s,
    const float* __restrict__ feat, float* __restrict__ rst, int dshift)
{
  const int e  = blockIdx.x;
  const int sN = src[e];
  const int dN = dst[e];
  __shared__ float alpha[HEADS];
  if (threadIdx.x < HEADS)
    alpha[threadIdx.x] = e_edge[e * HEADS + threadIdx.x] / s[dN * HEADS + threadIdx.x];
  __syncthreads();
  const int HD = HEADS << dshift;
  const float* fs = feat + (size_t)sN * HD;
  float*       rd = rst  + (size_t)dN * HD;
  for (int idx = threadIdx.x; idx < HD; idx += 256)
    atomicAdd(&rd[idx], alpha[idx >> dshift] * fs[idx]);
}

// out = relu(rst [+ res] + bias)
__global__ void k_finalize_relu(const float* __restrict__ rst, const float* __restrict__ res,
                                const float* __restrict__ bias, float* __restrict__ out,
                                size_t total, int hdMask)
{
  size_t i = (size_t)blockIdx.x * blockDim.x + threadIdx.x;
  if (i >= total) return;
  float v = rst[i] + bias[(int)(i & (size_t)hdMask)];
  if (res) v += res[i];
  out[i] = v > 0.f ? v : 0.f;
}

// d_out[n,o] = mean_h (rst[n,h,o] + b2[h,o])   (res2 already folded into rst)
__global__ void k_final_mean(const float* __restrict__ rst, const float* __restrict__ bias,
                             float* __restrict__ out, int Nn)
{
  const int i = blockIdx.x * blockDim.x + threadIdx.x;
  if (i >= Nn * OUTD) return;
  const int n = i / OUTD;
  const int o = i - n * OUTD;
  const float* r = rst + (size_t)n * (HEADS * OUTD);
  float acc = 0.f;
  #pragma unroll
  for (int h = 0; h < HEADS; ++h) acc += r[h * OUTD + o] + bias[h * OUTD + o];
  out[i] = acc * (1.0f / HEADS);
}

// ---------------------------------------------------------------------------
// host driver
// ---------------------------------------------------------------------------
static inline char* ws_take(char*& p, size_t bytes) {
  char* r = p;
  p += (bytes + 255) & ~(size_t)255;
  return r;
}
static inline int cdiv(size_t a, int b) { return (int)((a + b - 1) / b); }

static inline void fill_zero(float* p, size_t n, hipStream_t s) {
  k_fill_zero4<<<cdiv(n / 4, 256), 256, 0, s>>>((float4*)p, n / 4);
}
static inline void cvt_f16(const float* in, _Float16* out, size_t n, hipStream_t s) {
  k_cvt_f16x4<<<cdiv(n / 4, 256), 256, 0, s>>>((const float4*)in, (uint2*)out, n / 4);
}
static inline void cvt_w_t(const float* in, _Float16* out, int K, int N, hipStream_t s) {
  k_cvt_transpose_f16<<<cdiv((size_t)K * N, 256), 256, 0, s>>>(in, out, K, N);
}

extern "C" void kernel_launch(void* const* d_in, const int* in_sizes, int n_in,
                              void* d_out, int out_size, void* d_ws, size_t ws_size,
                              hipStream_t stream) {
  (void)n_in; (void)out_size; (void)ws_size;

  const float* x    = (const float*)d_in[0];
  const int*   src  = (const int*)d_in[1];
  const int*   dst  = (const int*)d_in[2];
  const float* W0   = (const float*)d_in[3];
  const float* al0  = (const float*)d_in[4];
  const float* ar0  = (const float*)d_in[5];
  const float* b0   = (const float*)d_in[6];
  const float* W1   = (const float*)d_in[7];
  const float* al1  = (const float*)d_in[8];
  const float* ar1  = (const float*)d_in[9];
  const float* b1   = (const float*)d_in[10];
  const float* W2   = (const float*)d_in[11];
  const float* al2  = (const float*)d_in[12];
  const float* ar2  = (const float*)d_in[13];
  const float* b2   = (const float*)d_in[14];
  const float* rW2  = (const float*)d_in[15];
  float* out = (float*)d_out;

  const int Nn = in_sizes[0] / HID;   // 20000
  const int Et = in_sizes[1];         // 340000 (random + self loops)
  const int HDmid = HEADS * HID;      // 512
  const int HDout = HEADS * OUTD;     // 1024

  // ---- workspace layout ----
  char* p = (char*)d_ws;
  _Float16* w0t  = (_Float16*)ws_take(p, (size_t)HID   * HDmid * 2);  // [N][K]
  _Float16* w1t  = (_Float16*)ws_take(p, (size_t)HDmid * HDmid * 2);
  _Float16* w2t  = (_Float16*)ws_take(p, (size_t)HDmid * HDout * 2);
  _Float16* rw2t = (_Float16*)ws_take(p, (size_t)HDmid * HDout * 2);
  _Float16* h16    = (_Float16*)ws_take(p, (size_t)Nn * HDmid * 2);
  float*    feat   = (float*)   ws_take(p, (size_t)Nn * HDout * 4);
  float*    rst    = (float*)   ws_take(p, (size_t)Nn * HDout * 4);
  float*    hbuf0  = (float*)   ws_take(p, (size_t)Nn * HDmid * 4);
  float*    hbuf1  = (float*)   ws_take(p, (size_t)Nn * HDmid * 4);
  float*    el     = (float*)   ws_take(p, (size_t)Nn * HEADS * 4);
  float*    er     = (float*)   ws_take(p, (size_t)Nn * HEADS * 4);
  unsigned* mOrd   = (unsigned*)ws_take(p, (size_t)Nn * HEADS * 4);
  float*    ssum   = (float*)   ws_take(p, (size_t)Nn * HEADS * 4);
  float*    e_edge = (float*)   ws_take(p, (size_t)Et * HEADS * 4);

  const int T = 256;

  // ---- convert + transpose weights to f16 [N][K] ----
  cvt_w_t(W0,  w0t,  HID,   HDmid, stream);
  cvt_w_t(W1,  w1t,  HDmid, HDmid, stream);
  cvt_w_t(W2,  w2t,  HDmid, HDout, stream);
  cvt_w_t(rW2, rw2t, HDmid, HDout, stream);

  const int gy = (Nn + TM - 1) / TM;
  const int ehBlocks = cdiv((size_t)Et * HEADS, T);
  const int nhBlocks = cdiv((size_t)Nn * HEADS * 32, T);

  // ================= layer 0: in HID, out HID/head, no res, relu ==========
  cvt_f16(x, h16, (size_t)Nn * HID, stream);
  k_gemm_wmma<<<dim3(HDmid / TN, gy), T, 0, stream>>>(h16, w0t, feat, Nn, HID, HDmid);
  fill_zero(rst, (size_t)Nn * HDmid, stream);
  fill_zero((float*)mOrd, (size_t)Nn * HEADS, stream);
  fill_zero(ssum, (size_t)Nn * HEADS, stream);
  k_attn_scores<<<nhBlocks, T, 0, stream>>>(feat, al0, ar0, el, er, Nn, HID);
  k_edge_max<<<ehBlocks, T, 0, stream>>>(src, dst, el, er, e_edge, mOrd, Et);
  k_edge_exp<<<ehBlocks, T, 0, stream>>>(dst, e_edge, mOrd, ssum, Et);
  k_edge_aggregate<<<Et, T, 0, stream>>>(src, dst, e_edge, ssum, feat, rst, 7);
  k_finalize_relu<<<cdiv((size_t)Nn * HDmid, T), T, 0, stream>>>(
      rst, nullptr, b0, hbuf0, (size_t)Nn * HDmid, HDmid - 1);

  // ================= layer 1: in 512, out HID/head, identity res, relu ====
  cvt_f16(hbuf0, h16, (size_t)Nn * HDmid, stream);
  k_gemm_wmma<<<dim3(HDmid / TN, gy), T, 0, stream>>>(h16, w1t, feat, Nn, HDmid, HDmid);
  fill_zero(rst, (size_t)Nn * HDmid, stream);
  fill_zero((float*)mOrd, (size_t)Nn * HEADS, stream);
  fill_zero(ssum, (size_t)Nn * HEADS, stream);
  k_attn_scores<<<nhBlocks, T, 0, stream>>>(feat, al1, ar1, el, er, Nn, HID);
  k_edge_max<<<ehBlocks, T, 0, stream>>>(src, dst, el, er, e_edge, mOrd, Et);
  k_edge_exp<<<ehBlocks, T, 0, stream>>>(dst, e_edge, mOrd, ssum, Et);
  k_edge_aggregate<<<Et, T, 0, stream>>>(src, dst, e_edge, ssum, feat, rst, 7);
  k_finalize_relu<<<cdiv((size_t)Nn * HDmid, T), T, 0, stream>>>(
      rst, hbuf0, b1, hbuf1, (size_t)Nn * HDmid, HDmid - 1);

  // ================= layer 2: in 512, out OUT/head, linear res, no act ====
  cvt_f16(hbuf1, h16, (size_t)Nn * HDmid, stream);
  k_gemm_wmma<<<dim3(HDout / TN, gy), T, 0, stream>>>(h16, w2t, feat, Nn, HDmid, HDout);
  // residual GEMM writes straight into rst -> acts as rst's initializer
  k_gemm_wmma<<<dim3(HDout / TN, gy), T, 0, stream>>>(h16, rw2t, rst, Nn, HDmid, HDout);
  fill_zero((float*)mOrd, (size_t)Nn * HEADS, stream);
  fill_zero(ssum, (size_t)Nn * HEADS, stream);
  k_attn_scores<<<nhBlocks, T, 0, stream>>>(feat, al2, ar2, el, er, Nn, OUTD);
  k_edge_max<<<ehBlocks, T, 0, stream>>>(src, dst, el, er, e_edge, mOrd, Et);
  k_edge_exp<<<ehBlocks, T, 0, stream>>>(dst, e_edge, mOrd, ssum, Et);
  k_edge_aggregate<<<Et, T, 0, stream>>>(src, dst, e_edge, ssum, feat, rst, 8);
  k_final_mean<<<cdiv((size_t)Nn * OUTD, T), T, 0, stream>>>(rst, b2, out, Nn);
}
